// MoELayer_40733469835921
// MI455X (gfx1250) — compile-verified, matrix operand order checked by reading, not run on hardware
//
#include <hip/hip_runtime.h>
#include <hip/hip_bf16.h>

// MoE SwiGLU layer for MI455X (gfx1250): wave32, WMMA bf16, TDM tensor loads
// for weight tiles (TENSORcnt), per-lane async-to-LDS for activation tiles
// (ASYNCcnt). T=8192 tokens, D=1024, E=8, HIDDEN=2730 (padded 2752), top-2.
//
//   zero(out) ; router -> combine[T,8] ; cast x,w1,w2,w3 -> padded bf16 in ws;
//   for e in 0..7: ffn1(e): H = silu(x@w1[e]) * (x@w2[e])   (bf16, ws, padded)
//                  ffn2(e): out += combine[:,e] * (H @ w3[e])
//
// Weight copies are K-MAJOR so WMMA fragments are 2x ds_load_b128, and every
// staged tile is a dense 2D block: B tiles move via ONE tensor_load_to_lds
// per block per K-step (TDM pads each 64B row to 80B in LDS = bank-friendly
// TS stride); A tiles via guard-free global_load_async_to_lds_b128.
//
// ws layout (bytes), total ~197.4 MB:
//   combine f32 : [0,           262144)
//   xb bf16     : [262144,      17039360)   [T][D]
//   w1t bf16    : [17039360,    62128128)   [E][HIDP][D]  (transposed, pad rows 0)
//   w2t bf16    : [62128128,    107216896)  [E][HIDP][D]  (transposed, pad rows 0)
//   w3t bf16    : [107216896,   152305664)  [E][D][HIDP]  (transposed, pad cols 0)
//   hbuf bf16   : [152305664,   197394432)  [T][HIDP]     (pad cols 0)

#define T_TOK 8192
#define DIM   1024
#define NEXP  8
#define HID   2730
#define HIDP  2752
#define TS    40   // LDS tile row stride (elements): 80B = 20 banks, conflict-free

typedef __bf16 bf16;
typedef __attribute__((ext_vector_type(16))) __bf16 v16bf;
typedef __attribute__((ext_vector_type(8)))  float  v8f;
typedef __attribute__((ext_vector_type(4)))  unsigned int su4;
typedef __attribute__((ext_vector_type(8)))  unsigned int su8;

// ---- CDNA5 async global->LDS (per-lane, ASYNCcnt) -------------------------
__device__ __forceinline__ void async_copy_b128(const bf16* gsrc, bf16* lds_dst) {
  const unsigned lds_off = (unsigned)(size_t)(void*)lds_dst;  // addr[31:0] = LDS offset
  asm volatile("global_load_async_to_lds_b128 %0, %1, off"
               :: "v"(lds_off), "v"((unsigned long long)(size_t)gsrc)
               : "memory");
}
__device__ __forceinline__ void async_wait0() { asm volatile("s_wait_asynccnt 0x0" ::: "memory"); }
__device__ __forceinline__ void async_wait1() { asm volatile("s_wait_asynccnt 0x1" ::: "memory"); }

// ---- CDNA5 Tensor Data Mover (TENSORcnt) ----------------------------------
// One 2D tile: tile_dim1=64 rows x tile_dim0=32 bf16 (64B), row stride `ld`
// elements in memory; LDS rows padded 64B->80B (pad_interval=16DW, amount=4DW)
// to match the TS=40 fragment layout. D# per CDNA5 ISA.
__device__ __forceinline__ void tdm_load_tile(const bf16* gsrc, const bf16* lds_dst,
                                              unsigned ld) {
  const unsigned long long ga = (unsigned long long)(size_t)gsrc;
  su4 g0;
  g0[0] = 1u;                                                 // count=1, user D#
  g0[1] = (unsigned)(size_t)(void*)lds_dst;                   // lds_addr
  g0[2] = (unsigned)ga;                                       // global_addr[31:0]
  g0[3] = (unsigned)((ga >> 32) & 0x01ffffffu) | (2u << 30);  // [56:32] | type=2
  su8 g1;
  g1[0] = (1u << 16) | (1u << 20) | (3u << 22) | (3u << 25);  // 2B, pad 4DW/16DW
  g1[1] = (ld & 0xffffu) << 16;                               // tensor_dim0 = ld
  g1[2] = 0xffffu << 16;                                      // tensor_dim1 = 0xffff
  g1[3] = 32u << 16;                                          // tile_dim0 = 32
  g1[4] = 64u;                                                // tile_dim1 = 64
  g1[5] = ld;                                                 // dim0_stride = ld
  g1[6] = 0u;
  g1[7] = 0u;
  asm volatile("tensor_load_to_lds %0, %1" :: "s"(g0), "s"(g1) : "memory");
}

// ---- WMMA fragment load (wave32 layout per CDNA5 ISA 7.12.2) --------------
// Tile in LDS is [rows][TS] with K contiguous. Lane (l<16): row rbase+l,
// K = {0..7, 16..23}; lane>=16: same rows, K = {8..15, 24..31}.
__device__ __forceinline__ v16bf load_frag(const bf16* tile, int rbase, int lane) {
  const int row = rbase + (lane & 15);
  const int ko  = (lane >> 4) ? 8 : 0;
  const bf16* p = tile + row * TS + ko;
  struct P { uint4 lo, hi; } t;
  t.lo = *reinterpret_cast<const uint4*>(p);       // K = ko .. ko+7
  t.hi = *reinterpret_cast<const uint4*>(p + 16);  // K = ko+16 .. ko+23
  return __builtin_bit_cast(v16bf, t);
}

__device__ __forceinline__ v8f wmma_bf16f32(v16bf a, v16bf b, v8f c) {
  return __builtin_amdgcn_wmma_f32_16x16x32_bf16(false, a, false, b, (short)0, c,
                                                 false, false);
}

// ---- utility kernels ------------------------------------------------------
__global__ __launch_bounds__(256) void zero_f32(float* __restrict__ p, int n4) {
  const int i = blockIdx.x * 256 + threadIdx.x;
  if (i < n4) reinterpret_cast<float4*>(p)[i] = make_float4(0.f, 0.f, 0.f, 0.f);
}

__global__ __launch_bounds__(256) void cast_f32_bf16(const float* __restrict__ in,
                                                     bf16* __restrict__ o, int n4) {
  const int i = blockIdx.x * 256 + threadIdx.x;
  if (i < n4) {
    const float4 f = reinterpret_cast<const float4*>(in)[i];
    bf16* p = o + (size_t)i * 4;
    p[0] = (bf16)f.x; p[1] = (bf16)f.y; p[2] = (bf16)f.z; p[3] = (bf16)f.w;
  }
}

// w1/w2: [E][D][HID] f32 -> transposed [E][HIDP][D] bf16 (pad rows zero).
__global__ __launch_bounds__(256) void cast_w12_t(const float* __restrict__ in,
                                                  bf16* __restrict__ o, int nquad) {
  const int i = blockIdx.x * 256 + threadIdx.x;
  if (i >= nquad) return;
  const size_t ob = (size_t)i * 4;
  const int d = (int)(ob % DIM);
  const int h = (int)((ob / DIM) % HIDP);
  const int e = (int)(ob / ((size_t)DIM * HIDP));
  bf16 v0 = (bf16)0.f, v1 = (bf16)0.f, v2 = (bf16)0.f, v3 = (bf16)0.f;
  if (h < HID) {
    const float* p = in + ((size_t)(e * DIM + d)) * HID + h;
    v0 = (bf16)p[0];
    v1 = (bf16)p[(size_t)HID];
    v2 = (bf16)p[(size_t)2 * HID];
    v3 = (bf16)p[(size_t)3 * HID];
  }
  bf16* q = o + ob;
  q[0] = v0; q[1] = v1; q[2] = v2; q[3] = v3;
}

// w3: [E][HID][D] f32 -> transposed [E][D][HIDP] bf16 (pad cols zero).
__global__ __launch_bounds__(256) void cast_w3_t(const float* __restrict__ in,
                                                 bf16* __restrict__ o, int npair) {
  const int i = blockIdx.x * 256 + threadIdx.x;
  if (i >= npair) return;
  const size_t op = (size_t)i * 2;
  const int h = (int)(op % HIDP);
  const int d = (int)((op / HIDP) % DIM);
  const int e = (int)(op / ((size_t)HIDP * DIM));
  bf16 a = (bf16)0.f, b = (bf16)0.f;
  if (h < HID) {  // HID even -> h+1 < HID too
    const float* p = in + ((size_t)e * HID + h) * DIM + d;
    a = (bf16)p[0];
    b = (bf16)p[(size_t)DIM];
  }
  bf16* q = o + op;
  q[0] = a; q[1] = b;
}

// ---- router: logits -> top-2 softmax -> combine[T, NEXP] ------------------
__global__ __launch_bounds__(256) void moe_router(const float* __restrict__ x,
                                                  const float* __restrict__ rw,
                                                  float* __restrict__ combine) {
  __shared__ float srw[DIM * NEXP];  // 32 KB
  for (int i = threadIdx.x; i < DIM * NEXP; i += 256) srw[i] = rw[i];
  __syncthreads();

  const int t = blockIdx.x * 256 + threadIdx.x;
  const float* xt = x + (size_t)t * DIM;
  float acc[NEXP];
#pragma unroll
  for (int e = 0; e < NEXP; ++e) acc[e] = 0.f;
  for (int d = 0; d < DIM; ++d) {
    const float xv = xt[d];
#pragma unroll
    for (int e = 0; e < NEXP; ++e) acc[e] += xv * srw[d * NEXP + e];
  }
  float v1 = -__builtin_inff(), v2 = -__builtin_inff();
  int i1 = 0, i2 = 0;
#pragma unroll
  for (int e = 0; e < NEXP; ++e) {
    const float v = acc[e];
    if (v > v1)      { v2 = v1; i2 = i1; v1 = v; i1 = e; }
    else if (v > v2) { v2 = v; i2 = e; }
  }
  const float p1 = 1.0f / (1.0f + __expf(v2 - v1));
  const float p2 = 1.0f - p1;
#pragma unroll
  for (int e = 0; e < NEXP; ++e)
    combine[(size_t)t * NEXP + e] = (e == i1) ? p1 : ((e == i2) ? p2 : 0.0f);
}

// ---- GEMM-1 (fused SwiGLU): H[t,h] = silu(x@w1) * (x@w2), bf16 ------------
// 64x64 tile/block, 8 waves; K-step 32; A via async, B1/B2 via TDM; peeled
// last iteration keeps the steady-state body branch-free.
__global__ __launch_bounds__(256) void moe_ffn1(const bf16* __restrict__ xb,
                                                const bf16* __restrict__ w1t,
                                                const bf16* __restrict__ w2t,
                                                bf16* __restrict__ hbuf) {
  __shared__ __align__(16) bf16 sA [2][64 * TS];
  __shared__ __align__(16) bf16 sB1[2][64 * TS];
  __shared__ __align__(16) bf16 sB2[2][64 * TS];

  const int tid  = threadIdx.x;
  const int lane = tid & 31;
  const int wave = tid >> 5;
  const int nt   = wave & 3;
  const int mt0  = (wave >> 2) * 2;
  const int h0   = blockIdx.x * 64;   // within [0, HIDP)
  const int t0   = blockIdx.y * 64;

  v8f c1[2], c2[2];
#pragma unroll
  for (int i = 0; i < 2; ++i)
#pragma unroll
    for (int v = 0; v < 8; ++v) { c1[i][v] = 0.f; c2[i][v] = 0.f; }

  const int crow = tid >> 2, ck8 = (tid & 3) * 8;  // A coop: 64 rows x 32 K
  const int lofs = crow * TS + ck8;

  const bf16* gA   = &xb [(size_t)(t0 + crow) * DIM + ck8];  // per-lane
  const bf16* gB1u = w1t + (size_t)h0 * DIM;                 // tile base (uniform)
  const bf16* gB2u = w2t + (size_t)h0 * DIM;

  auto compute = [&](int cur) {
    const v16bf b1 = load_frag(&sB1[cur][0], nt * 16, lane);
    const v16bf b2 = load_frag(&sB2[cur][0], nt * 16, lane);
#pragma unroll
    for (int i = 0; i < 2; ++i) {
      const v16bf a = load_frag(&sA[cur][0], (mt0 + i) * 16, lane);
      c1[i] = wmma_bf16f32(a, b1, c1[i]);
      c2[i] = wmma_bf16f32(a, b2, c2[i]);
    }
  };

  async_copy_b128(gA, &sA[0][lofs]);
  if (wave == 0) {
    tdm_load_tile(gB1u, &sB1[0][0], DIM);
    tdm_load_tile(gB2u, &sB2[0][0], DIM);
  }

  const int KT = DIM / 32;  // 32
  for (int kk = 0; kk < KT - 1; ++kk) {
    const int cur = kk & 1, nxt = cur ^ 1;
    const int k1 = (kk + 1) * 32;
    async_copy_b128(gA + k1, &sA[nxt][lofs]);
    if (wave == 0) {
      tdm_load_tile(gB1u + k1, &sB1[nxt][0], DIM);
      tdm_load_tile(gB2u + k1, &sB2[nxt][0], DIM);
    }
    async_wait1();                                  // my A tile landed
    if (wave == 0) __builtin_amdgcn_s_wait_tensorcnt(2);  // current B tiles landed
    __syncthreads();
    compute(cur);
    __syncthreads();                                // all done before refill
  }
  async_wait0();
  if (wave == 0) __builtin_amdgcn_s_wait_tensorcnt(0);
  __syncthreads();
  compute((KT - 1) & 1);

  // epilogue: silu*gate; pad cols [HID,HIDP) written as zeros
  const int hi = lane >> 4;
  const int hc = h0 + nt * 16 + (lane & 15);
#pragma unroll
  for (int i = 0; i < 2; ++i)
#pragma unroll
    for (int v = 0; v < 8; ++v) {
      const int row = t0 + (mt0 + i) * 16 + v + 8 * hi;
      float val = 0.f;
      if (hc < HID) {
        const float g = c1[i][v];
        val = (g / (1.0f + __expf(-g))) * c2[i][v];
      }
      hbuf[(size_t)row * HIDP + hc] = (bf16)val;
    }
}

// ---- GEMM-2: out[t,d] += combine[t,e] * (H @ w3[e]) -----------------------
__global__ __launch_bounds__(256) void moe_ffn2(const bf16* __restrict__ hbuf,
                                                const bf16* __restrict__ w3t,
                                                const float* __restrict__ combine,
                                                float* __restrict__ out, int e) {
  __shared__ __align__(16) bf16 sA[2][64 * TS];
  __shared__ __align__(16) bf16 sB[2][64 * TS];

  const int tid  = threadIdx.x;
  const int lane = tid & 31;
  const int wave = tid >> 5;
  const int nt   = wave & 3;
  const int mt0  = (wave >> 2) * 2;
  const int d0   = blockIdx.x * 64;
  const int t0   = blockIdx.y * 64;

  v8f c[2];
#pragma unroll
  for (int i = 0; i < 2; ++i)
#pragma unroll
    for (int v = 0; v < 8; ++v) c[i][v] = 0.f;

  const int crow = tid >> 2, ck8 = (tid & 3) * 8;
  const int lofs = crow * TS + ck8;

  const bf16* gA  = &hbuf[(size_t)(t0 + crow) * HIDP + ck8];  // per-lane
  const bf16* gBu = w3t + (size_t)d0 * HIDP;                  // tile base

  auto compute = [&](int cur) {
    const v16bf b = load_frag(&sB[cur][0], nt * 16, lane);
#pragma unroll
    for (int i = 0; i < 2; ++i) {
      const v16bf a = load_frag(&sA[cur][0], (mt0 + i) * 16, lane);
      c[i] = wmma_bf16f32(a, b, c[i]);
    }
  };

  async_copy_b128(gA, &sA[0][lofs]);
  if (wave == 0) tdm_load_tile(gBu, &sB[0][0], HIDP);

  const int KT = HIDP / 32;  // 86, guard-free (padded K, zeroed pads)
  for (int kk = 0; kk < KT - 1; ++kk) {
    const int cur = kk & 1, nxt = cur ^ 1;
    const int k1 = (kk + 1) * 32;
    async_copy_b128(gA + k1, &sA[nxt][lofs]);
    if (wave == 0) tdm_load_tile(gBu + k1, &sB[nxt][0], HIDP);
    async_wait1();
    if (wave == 0) __builtin_amdgcn_s_wait_tensorcnt(1);
    __syncthreads();
    compute(cur);
    __syncthreads();
  }
  async_wait0();
  if (wave == 0) __builtin_amdgcn_s_wait_tensorcnt(0);
  __syncthreads();
  compute((KT - 1) & 1);

  // epilogue: scale by routing weight, accumulate (experts run serially)
  const int hi  = lane >> 4;
  const int col = d0 + nt * 16 + (lane & 15);
#pragma unroll
  for (int i = 0; i < 2; ++i)
#pragma unroll
    for (int v = 0; v < 8; ++v) {
      const int row = t0 + (mt0 + i) * 16 + v + 8 * hi;
      const float w = combine[(size_t)row * NEXP + e];
      out[(size_t)row * DIM + col] += w * c[i][v];
    }
}

// ---- host-side launch -----------------------------------------------------
extern "C" void kernel_launch(void* const* d_in, const int* in_sizes, int n_in,
                              void* d_out, int out_size, void* d_ws, size_t ws_size,
                              hipStream_t stream) {
  (void)in_sizes; (void)n_in; (void)out_size; (void)ws_size;
  const float* x  = (const float*)d_in[0];
  const float* rw = (const float*)d_in[1];
  const float* w1 = (const float*)d_in[2];
  const float* w2 = (const float*)d_in[3];
  const float* w3 = (const float*)d_in[4];
  float* out = (float*)d_out;

  char* ws = (char*)d_ws;
  float* combine = (float*)(ws + 0);
  bf16*  xb   = (bf16*)(ws + 262144);
  bf16*  w1t  = (bf16*)(ws + 17039360);
  bf16*  w2t  = (bf16*)(ws + 62128128);
  bf16*  w3t  = (bf16*)(ws + 107216896);
  bf16*  hbuf = (bf16*)(ws + 152305664);

  const int XN4   = T_TOK * DIM / 4;          // 2,097,152 quads
  const int NQUAD = NEXP * HIDP * DIM / 4;    // 5,636,096 quads (w1/w2 out)
  const int NPAIR = NEXP * DIM * HIDP / 2;    // 11,272,192 pairs (w3 out)

  zero_f32<<<(XN4 + 255) / 256, 256, 0, stream>>>(out, XN4);
  moe_router<<<T_TOK / 256, 256, 0, stream>>>(x, rw, combine);
  cast_f32_bf16<<<(XN4 + 255) / 256, 256, 0, stream>>>(x, xb, XN4);
  cast_w12_t<<<(NQUAD + 255) / 256, 256, 0, stream>>>(w1, w1t, NQUAD);
  cast_w12_t<<<(NQUAD + 255) / 256, 256, 0, stream>>>(w2, w2t, NQUAD);
  cast_w3_t<<<(NPAIR + 255) / 256, 256, 0, stream>>>(w3, w3t, NPAIR);

  const dim3 g1(HIDP / 64, T_TOK / 64);  // 43 x 128
  const dim3 g2(DIM / 64,  T_TOK / 64);  // 16 x 128
  for (int e = 0; e < NEXP; ++e) {
    const size_t wo12 = (size_t)e * HIDP * DIM;
    const size_t wo3  = (size_t)e * DIM * HIDP;
    moe_ffn1<<<g1, 256, 0, stream>>>(xb, w1t + wo12, w2t + wo12, hbuf);
    moe_ffn2<<<g2, 256, 0, stream>>>(hbuf, w3t + wo3, combine, out, e);
  }
}
